// Discriminator_1795296330384
// MI455X (gfx1250) — matrix operand order repaired and checked by reading, not compiled, not run on hardware
//
#include <hip/hip_runtime.h>

// ---------------------------------------------------------------------------
// Fused discriminator forward for MI455X (gfx1250, wave32, WMMA + TDM).
// All GEMMs -> v_wmma_f32_16x16x32_f16. Conv folded into weights analytically.
// Weight staging global->LDS via Tensor Data Mover (tensor_load_to_lds,
// TENSORcnt). One workgroup per 128-row tile; activations live in LDS.
// ---------------------------------------------------------------------------

typedef __attribute__((ext_vector_type(16))) _Float16 v16h;
typedef __attribute__((ext_vector_type(8)))  float    v8f;
typedef __attribute__((ext_vector_type(4)))  float    f32x4;
typedef unsigned int u32x4 __attribute__((ext_vector_type(4)));
typedef int          i32x8 __attribute__((ext_vector_type(8)));
typedef int          i32x4 __attribute__((ext_vector_type(4)));

constexpr int B_ROWS = 32768;
constexpr int D_IN   = 128;
constexpr int H      = 256;
constexpr int NB     = 4;
constexpr int TM     = 128;   // rows per workgroup
constexpr int NTH    = 256;   // 8 waves of 32

// LDS strides (elements) -- padded to dodge bank conflicts, keep 16B alignment
constexpr int ASTR = 264;     // f32 activation stride
constexpr int HSTR = 264;     // f16 scratch stride
constexpr int WS1  = 136;     // staged W1 stride   (K=128)
constexpr int WSW  = 264;     // staged 256-K stride
constexpr int WSC  = 104;     // staged Wcat stride (K=96)

// workspace layout (in _Float16 elements) -- rows PRE-PADDED to LDS stride so
// every stage is a contiguous 1-D TDM copy.
constexpr size_t OFF_W1   = 0;                                   // 256 x WS1
constexpr size_t OFF_WE   = OFF_W1   + (size_t)H * WS1;          // NB x 96 x WSW
constexpr size_t OFF_WCAT = OFF_WE   + (size_t)NB * 96 * WSW;    // NB x 256 x WSC
constexpr size_t OFF_WF1  = OFF_WCAT + (size_t)NB * H * WSC;     // NB x 256 x WSW
constexpr size_t OFF_WF2  = OFF_WF1  + (size_t)NB * H * WSW;     // NB x 256 x WSW
constexpr size_t WS_HALVES = OFF_WF2 + (size_t)NB * H * WSW;
constexpr size_t OFF_BCC_BYTES = WS_HALVES * 2;                  // NB x 96 f32

// LDS layout (bytes)
constexpr int SM_ACT = TM * ASTR * 4;           // 135168
constexpr int SM_W   = 256 * WS1 * 2;           // 69632 (max staged shape)
constexpr int SM_H16 = TM * HSTR * 2;           // 67584
constexpr int SMEM_BYTES = SM_ACT + SM_W + SM_H16;   // 272384 < 320KB/WGP

union F16Frag { v16h v; f32x4 q[2]; _Float16 h[16]; };

// A fragment (16x32 f16): lane m = lane&15; K chunks at k0+hi*8 and k0+16+hi*8
__device__ __forceinline__ v16h load_a16(const _Float16* buf, int stride, int m,
                                         int k0, int hi) {
  const _Float16* p = buf + m * stride + k0 + hi * 8;
  F16Frag a;
  a.q[0] = *(const f32x4*)(p);
  a.q[1] = *(const f32x4*)(p + 16);
  return a.v;
}

__device__ __forceinline__ v16h load_a32(const float* buf, int stride, int m,
                                         int k0, int hi) {
  const float* p = buf + m * stride + k0 + hi * 8;
  f32x4 c0 = *(const f32x4*)(p);
  f32x4 c1 = *(const f32x4*)(p + 4);
  f32x4 c2 = *(const f32x4*)(p + 16);
  f32x4 c3 = *(const f32x4*)(p + 20);
  F16Frag a;
#pragma unroll
  for (int j = 0; j < 4; ++j) {
    a.h[j]      = (_Float16)c0[j];
    a.h[4 + j]  = (_Float16)c1[j];
    a.h[8 + j]  = (_Float16)c2[j];
    a.h[12 + j] = (_Float16)c3[j];
  }
  return a.v;
}

// B fragment (32x16 f16): lane n = lane&15; K chunks at k0+hi*8 and k0+16+hi*8
__device__ __forceinline__ v16h load_b(const _Float16* wb, int stride, int n,
                                       int k0, int hi) {
  const _Float16* p = wb + n * stride + k0 + hi * 8;
  F16Frag b;
  b.q[0] = *(const f32x4*)(p);
  b.q[1] = *(const f32x4*)(p + 16);
  return b.v;
}

// ---- Tensor Data Mover: contiguous 1-D copy of nhalves f16 global->LDS ----
// D# per CDNA5 ISA 8.3/8.4: group0 = {count=1 | lds_addr | global_addr | type=2}
// group1 = {data_size=1(2B), tensor_dim0=tile_dim0=nhalves, tensor_dim1=1,
//           tensor_dim0_stride=nhalves}. Higher-dim groups zero (1-D copy).
// Issued once by the calling wave; completion tracked with TENSORcnt.
__device__ __forceinline__ void tdm_stage(const _Float16* gsrc, _Float16* ldst,
                                          unsigned nhalves) {
  unsigned long long ga = (unsigned long long)(size_t)gsrc;
  unsigned lds = (unsigned)(size_t)ldst;   // LDS aperture: low 32 bits = offset
  u32x4 g0;
  g0[0] = 1u;                                             // count=1, user desc
  g0[1] = lds;                                            // lds_addr
  g0[2] = (unsigned)ga;                                   // global_addr[31:0]
  g0[3] = (unsigned)((ga >> 32) & 0x01FFFFFFull) | 0x80000000u; // [56:32]|type=2
  unsigned nlo = nhalves & 0xFFFFu, nhi = (nhalves >> 16) & 0xFFFFu;
  i32x8 g1;
  g1[0] = (int)(1u << 16);            // workgroup_mask=0, data_size=1 (2 bytes)
  g1[1] = (int)(nlo << 16);           // tensor_dim0[15:0]
  g1[2] = (int)(nhi | (1u << 16));    // tensor_dim0[31:16], tensor_dim1=1
  g1[3] = (int)(nlo << 16);           // tile_dim0 = nhalves
  g1[4] = 0;                          // tile_dim1=0 (1-D), tile_dim2=0
  g1[5] = (int)nhalves;               // tensor_dim0_stride[31:0]
  g1[6] = 0;
  g1[7] = 0;
  i32x4 z4 = {0, 0, 0, 0};
  i32x8 z8 = {0, 0, 0, 0, 0, 0, 0, 0};
  __builtin_amdgcn_tensor_load_to_lds(g0, g1, z4, z4, z8, 0);
  __builtin_amdgcn_s_wait_tensorcnt(0);
}

// barrier-wrapped stage: wave 0 issues the DMA, whole workgroup syncs
#define STAGE(gsrc, nhalves)                       \
  do {                                             \
    __syncthreads();                               \
    if (wv == 0) tdm_stage((gsrc), wbuf, (nhalves)); \
    __syncthreads();                               \
  } while (0)

template <int KSTEPS, int NT, bool A32>
__device__ __forceinline__ void mma_block(const float* a32, const _Float16* a16,
                                          int astride, int am, int hi, int lanelo,
                                          const _Float16* wb, int wstride, v8f* acc) {
#pragma unroll
  for (int ks = 0; ks < KSTEPS; ++ks) {
    v16h a;
    if constexpr (A32) a = load_a32(a32, astride, am, ks * 32, hi);
    else               a = load_a16(a16, astride, am, ks * 32, hi);
#pragma unroll
    for (int t = 0; t < NT; ++t) {
      v16h b = load_b(wb, wstride, t * 16 + lanelo, ks * 32, hi);
      acc[t] = __builtin_amdgcn_wmma_f32_16x16x32_f16(
          false, a, false, b, (short)0, acc[t], false, false);
    }
  }
}

// LayerNorm over a wave's 16 rows x 256 cols held in acc[16] (C-layout:
// element (m = m0 + r + 8*hi, n = t*16 + lanelo)), result -> f32 LDS.
__device__ __forceinline__ void ln_rows(v8f acc[16], float* act, int m0, int hi,
                                        int lanelo, const float* g,
                                        const float* be) {
  float s1[8], s2[8];
#pragma unroll
  for (int r = 0; r < 8; ++r) { s1[r] = 0.f; s2[r] = 0.f; }
#pragma unroll
  for (int t = 0; t < 16; ++t)
#pragma unroll
    for (int r = 0; r < 8; ++r) {
      float v = acc[t][r];
      s1[r] += v; s2[r] += v * v;
    }
#pragma unroll
  for (int stp = 0; stp < 4; ++stp) {     // 16-lane butterfly (stays in hi half)
    int msk = 8 >> stp;
#pragma unroll
    for (int r = 0; r < 8; ++r) {
      s1[r] += __shfl_xor(s1[r], msk, 32);
      s2[r] += __shfl_xor(s2[r], msk, 32);
    }
  }
  float mean[8], rstd[8];
#pragma unroll
  for (int r = 0; r < 8; ++r) {
    float m = s1[r] * (1.f / 256.f);
    float v = s2[r] * (1.f / 256.f) - m * m;
    mean[r] = m;
    rstd[r] = rsqrtf(v + 1e-5f);
  }
#pragma unroll
  for (int t = 0; t < 16; ++t) {
    int n = t * 16 + lanelo;
    float gv = g[n], bv = be[n];
#pragma unroll
    for (int r = 0; r < 8; ++r) {
      int m = m0 + r + 8 * hi;
      act[m * ASTR + n] = (acc[t][r] - mean[r]) * rstd[r] * gv + bv;
    }
  }
}

__global__ __launch_bounds__(NTH, 1)
void fused_disc(const float* __restrict__ x, const _Float16* __restrict__ wsh,
                const float* __restrict__ bcc, const float* __restrict__ b1,
                const float* __restrict__ a0p, const float* __restrict__ g0,
                const float* __restrict__ be0, const float* __restrict__ bcat,
                const float* __restrict__ g1, const float* __restrict__ be1,
                const float* __restrict__ bf1, const float* __restrict__ afp,
                const float* __restrict__ bf2, const float* __restrict__ g2,
                const float* __restrict__ be2, const float* __restrict__ Wout,
                const float* __restrict__ boutp, float* __restrict__ out) {
  extern __shared__ char smem[];
  float*    act32 = (float*)smem;
  _Float16* wbuf  = (_Float16*)(smem + SM_ACT);
  _Float16* h16   = (_Float16*)(smem + SM_ACT + SM_W);

  const int tid = threadIdx.x;
  const int lane = tid & 31, lanelo = lane & 15, hi = lane >> 4;
  const int wv = tid >> 5;
  const int m0 = wv * 16;          // this wave's 16-row slice (rows are private)
  const int am = m0 + lanelo;      // A-fragment row for this lane
  const size_t rowbase = (size_t)blockIdx.x * TM;

  // load x tile (128 rows x 128 f32) into act32
  {
    const float* xg = x + rowbase * D_IN;
    for (int i = tid; i < TM * (D_IN / 4); i += NTH) {
      int r = i >> 5;
      int c = (i & 31) << 2;
      *(f32x4*)(act32 + r * ASTR + c) = *(const f32x4*)(xg + r * D_IN + c);
    }
  }

  const float a0 = a0p[0];
  const v8f vzero = {0.f, 0.f, 0.f, 0.f, 0.f, 0.f, 0.f, 0.f};
  v8f acc[16];

  // ---- GEMM1: prelu(x @ W1.T + b1) then LN ----
  STAGE(wsh + OFF_W1, (unsigned)(H * WS1));
#pragma unroll
  for (int t = 0; t < 16; ++t) acc[t] = vzero;
  mma_block<4, 16, true>(act32, nullptr, ASTR, am, hi, lanelo, wbuf, WS1, acc);
#pragma unroll
  for (int t = 0; t < 16; ++t) {
    float bv = b1[t * 16 + lanelo];
#pragma unroll
    for (int r = 0; r < 8; ++r) {
      float v = acc[t][r] + bv;
      acc[t][r] = v >= 0.f ? v : a0 * v;
    }
  }
  ln_rows(acc, act32, m0, hi, lanelo, g0, be0);

  for (int ib = 0; ib < NB; ++ib) {
    // ---- conv folded into weights: y = lrelu(out @ We.T + bcc, 0.1) ----
    STAGE(wsh + OFF_WE + (size_t)ib * 96 * WSW, (unsigned)(96 * WSW));
    v8f accB[6];
#pragma unroll
    for (int t = 0; t < 6; ++t) accB[t] = vzero;
    mma_block<8, 6, true>(act32, nullptr, ASTR, am, hi, lanelo, wbuf, WSW, accB);
#pragma unroll
    for (int t = 0; t < 6; ++t) {
      int n = t * 16 + lanelo;
      float bv = bcc[ib * 96 + n];
#pragma unroll
      for (int r = 0; r < 8; ++r) {
        float v = accB[t][r] + bv;
        v = v >= 0.f ? v : 0.1f * v;
        h16[(m0 + r + 8 * hi) * HSTR + n] = (_Float16)v;
      }
    }

    // ---- concat GEMM: out = ln(y @ Wcat.T + bcat + res) ----
    STAGE(wsh + OFF_WCAT + (size_t)ib * H * WSC, (unsigned)(H * WSC));
#pragma unroll
    for (int t = 0; t < 16; ++t) acc[t] = vzero;
    mma_block<3, 16, false>(nullptr, h16, HSTR, am, hi, lanelo, wbuf, WSC, acc);
#pragma unroll
    for (int t = 0; t < 16; ++t) {
      int n = t * 16 + lanelo;
      float bv = bcat[ib * H + n];
#pragma unroll
      for (int r = 0; r < 8; ++r) {
        int m = m0 + r + 8 * hi;
        acc[t][r] += bv + act32[m * ASTR + n];   // residual (f32 from LDS)
      }
    }
    ln_rows(acc, act32, m0, hi, lanelo, g1 + ib * H, be1 + ib * H);

    // ---- FFN1: h = prelu(out @ Wf1.T + bf1, af) ---- (weights in 2 N-halves)
    const _Float16* wf1 = wsh + OFF_WF1 + (size_t)ib * H * WSW;
#pragma unroll
    for (int t = 0; t < 16; ++t) acc[t] = vzero;
    STAGE(wf1, (unsigned)(128 * WSW));
    mma_block<8, 8, true>(act32, nullptr, ASTR, am, hi, lanelo, wbuf, WSW, acc);
    STAGE(wf1 + 128 * WSW, (unsigned)(128 * WSW));
    mma_block<8, 8, true>(act32, nullptr, ASTR, am, hi, lanelo, wbuf, WSW, acc + 8);
    {
      float afi = afp[ib];
#pragma unroll
      for (int t = 0; t < 16; ++t) {
        int n = t * 16 + lanelo;
        float bv = bf1[ib * H + n];
#pragma unroll
        for (int r = 0; r < 8; ++r) {
          float v = acc[t][r] + bv;
          v = v >= 0.f ? v : afi * v;
          h16[(m0 + r + 8 * hi) * HSTR + n] = (_Float16)v;
        }
      }
    }

    // ---- FFN2: out = ln(h @ Wf2.T + bf2 + res) ----
    const _Float16* wf2 = wsh + OFF_WF2 + (size_t)ib * H * WSW;
#pragma unroll
    for (int t = 0; t < 16; ++t) acc[t] = vzero;
    STAGE(wf2, (unsigned)(128 * WSW));
    mma_block<8, 8, false>(nullptr, h16, HSTR, am, hi, lanelo, wbuf, WSW, acc);
    STAGE(wf2 + 128 * WSW, (unsigned)(128 * WSW));
    mma_block<8, 8, false>(nullptr, h16, HSTR, am, hi, lanelo, wbuf, WSW, acc + 8);
#pragma unroll
    for (int t = 0; t < 16; ++t) {
      int n = t * 16 + lanelo;
      float bv = bf2[ib * H + n];
#pragma unroll
      for (int r = 0; r < 8; ++r) {
        int m = m0 + r + 8 * hi;
        acc[t][r] += bv + act32[m * ASTR + n];
      }
    }
    ln_rows(acc, act32, m0, hi, lanelo, g2 + ib * H, be2 + ib * H);
  }

  // ---- final projection: out @ Wout.T + bout (f32, row-private) ----
  {
    int row = m0 + lanelo;
    const float* ar = act32 + row * ASTR + hi * 128;
    const float* wr = Wout + hi * 128;
    float s = 0.f;
#pragma unroll 4
    for (int j = 0; j < 128; ++j) s += ar[j] * wr[j];
    s += __shfl_xor(s, 16, 32);
    if (hi == 0) out[rowbase + row] = s + boutp[0];
  }
}

// -------------------- weight prep kernels (run every call) ------------------
// Write f16 weights with rows PRE-PADDED to the LDS stage stride, so staging
// is a contiguous 1-D TDM copy. Pad columns are never read.

__global__ void k_cvt_pad(const float* __restrict__ s, _Float16* __restrict__ d,
                          int rows, int K, int stride) {
  int i = blockIdx.x * 256 + threadIdx.x;
  if (i >= rows * K) return;
  int r = i / K, c = i - r * K;
  d[r * stride + c] = (_Float16)s[i];
}

// Fold 25-tap Gaussian conv into the projection weights:
// We[j*32+c, k] = sum_t f_j[t] * Wc_j[c, k - t + 12]  (zero padded)
__global__ void k_weff(const float* __restrict__ filt, const float* __restrict__ Wc1,
                       const float* __restrict__ Wc2, const float* __restrict__ Wc3,
                       _Float16* __restrict__ We) {
  int i = blockIdx.x * 256 + threadIdx.x;
  if (i >= NB * 96 * H) return;
  int k = i & (H - 1);
  int rc = i >> 8;
  int row = rc % 96;
  int ib = rc / 96;
  int j = row / 32, c = row % 32;
  const float* Wc = (j == 0) ? Wc1 : (j == 1) ? Wc2 : Wc3;
  const float* wrow = Wc + ((size_t)ib * 32 + c) * H;
  const float* fr = filt + j * 25;
  float s = 0.f;
#pragma unroll
  for (int t = 0; t < 25; ++t) {
    int h = k + 12 - t;
    if (h >= 0 && h < H) s += fr[t] * wrow[h];
  }
  We[(size_t)rc * WSW + k] = (_Float16)s;   // padded row stride
}

__global__ void k_bcc(const float* __restrict__ bc1, const float* __restrict__ bc2,
                      const float* __restrict__ bc3, float* __restrict__ bcc) {
  int i = blockIdx.x * 256 + threadIdx.x;
  if (i >= NB * 96) return;
  int ib = i / 96, r = i % 96;
  int j = r / 32, c = r % 32;
  const float* b = (j == 0) ? bc1 : (j == 1) ? bc2 : bc3;
  bcc[i] = b[ib * 32 + c];
}

// ---------------------------------------------------------------------------

extern "C" void kernel_launch(void* const* d_in, const int* in_sizes, int n_in,
                              void* d_out, int out_size, void* d_ws, size_t ws_size,
                              hipStream_t stream) {
  (void)in_sizes; (void)n_in; (void)out_size; (void)ws_size;
  const float* x    = (const float*)d_in[0];
  const float* filt = (const float*)d_in[1];
  const float* W1   = (const float*)d_in[2];
  const float* b1   = (const float*)d_in[3];
  const float* a0   = (const float*)d_in[4];
  const float* g0   = (const float*)d_in[5];
  const float* be0  = (const float*)d_in[6];
  const float* Wc1  = (const float*)d_in[7];
  const float* bc1  = (const float*)d_in[8];
  const float* Wc2  = (const float*)d_in[9];
  const float* bc2  = (const float*)d_in[10];
  const float* Wc3  = (const float*)d_in[11];
  const float* bc3  = (const float*)d_in[12];
  const float* Wcat = (const float*)d_in[13];
  const float* bcat = (const float*)d_in[14];
  const float* g1   = (const float*)d_in[15];
  const float* be1  = (const float*)d_in[16];
  const float* Wf1  = (const float*)d_in[17];
  const float* bf1  = (const float*)d_in[18];
  const float* af   = (const float*)d_in[19];
  const float* Wf2  = (const float*)d_in[20];
  const float* bf2  = (const float*)d_in[21];
  const float* g2   = (const float*)d_in[22];
  const float* be2  = (const float*)d_in[23];
  const float* Wout = (const float*)d_in[24];
  const float* bout = (const float*)d_in[25];
  float* out = (float*)d_out;

  _Float16* ws16 = (_Float16*)d_ws;
  float* bcc = (float*)((char*)d_ws + OFF_BCC_BYTES);

  k_cvt_pad<<<(H * D_IN + 255) / 256, 256, 0, stream>>>(
      W1, ws16 + OFF_W1, H, D_IN, WS1);
  k_cvt_pad<<<(NB * H * 96 + 255) / 256, 256, 0, stream>>>(
      Wcat, ws16 + OFF_WCAT, NB * H, 96, WSC);
  k_cvt_pad<<<(NB * H * H + 255) / 256, 256, 0, stream>>>(
      Wf1, ws16 + OFF_WF1, NB * H, H, WSW);
  k_cvt_pad<<<(NB * H * H + 255) / 256, 256, 0, stream>>>(
      Wf2, ws16 + OFF_WF2, NB * H, H, WSW);
  k_weff<<<(NB * 96 * H + 255) / 256, 256, 0, stream>>>(
      filt, Wc1, Wc2, Wc3, ws16 + OFF_WE);
  k_bcc<<<2, 256, 0, stream>>>(bc1, bc2, bc3, bcc);

  (void)hipFuncSetAttribute(reinterpret_cast<const void*>(&fused_disc),
                            hipFuncAttributeMaxDynamicSharedMemorySize,
                            SMEM_BYTES);

  fused_disc<<<B_ROWS / TM, NTH, SMEM_BYTES, stream>>>(
      x, ws16, bcc, b1, a0, g0, be0, bcat, g1, be1, bf1, af, bf2, g2, be2,
      Wout, bout, out);
}
